// RoPEAttention_42004780155530
// MI455X (gfx1250) — matrix operand order repaired and checked by reading, not compiled, not run on hardware
//
#include <hip/hip_runtime.h>

// Problem constants (match the reference).
#define Bc  2
#define Nc  2048
#define Dc  1024
#define Hc  16
#define HDc 64

typedef __attribute__((ext_vector_type(16))) __bf16 v16bf;
typedef __attribute__((ext_vector_type(8)))  float  v8f;
typedef __attribute__((ext_vector_type(4)))  unsigned int u32x4;
typedef __attribute__((ext_vector_type(8)))  int          i32x8;
typedef __attribute__((ext_vector_type(4)))  int          i32x4;
typedef __attribute__((ext_vector_type(2)))  unsigned int u32x2;
typedef __attribute__((ext_vector_type(4)))  float        f32x4;

#if defined(__has_builtin)
#if __has_builtin(__builtin_amdgcn_tensor_load_to_lds)
#define HAVE_TDM 1
#else
#define HAVE_TDM 0
#endif
#else
#define HAVE_TDM 0
#endif

union Frag16 {
  v16bf  v;
  u32x4  q[2];
  __bf16 e[16];
};

// Load a 16-element bf16 fragment from two 8-element (16B) runs.
__device__ __forceinline__ v16bf load2x8(const __bf16* p0, const __bf16* p1) {
  Frag16 f;
  f.q[0] = *(const u32x4*)p0;
  f.q[1] = *(const u32x4*)p1;
  return f.v;
}

__device__ __forceinline__ v8f wmma_bf16(v16bf a, v16bf b, v8f c) {
  // (neg_a, A, neg_b, B, c_mod, C, reuse_a, reuse_b)
  return __builtin_amdgcn_wmma_f32_16x16x32_bf16(false, a, false, b, (short)0, c,
                                                 false, false);
}

// ---------------------------------------------------------------------------
// Kernel 0: weight transpose + f32->bf16 convert.  Wt[c][k] = (bf16)W[k][c].
// grid = (D/32, D/32, 4); block = 256.
// ---------------------------------------------------------------------------
__global__ __launch_bounds__(256)
void wtrans_kernel(const float* __restrict__ Wq, const float* __restrict__ Wk,
                   const float* __restrict__ Wv, const float* __restrict__ Wo,
                   unsigned short* __restrict__ wt) {
  __shared__ __align__(16) __bf16 T[32][40];   // 32x32 tile, padded rows
  const int tid = threadIdx.x;
  const int k0  = blockIdx.x * 32;             // source row block (K dim)
  const int c0  = blockIdx.y * 32;             // source col block
  const int z   = blockIdx.z;
  const float* W = (z == 0) ? Wq : (z == 1) ? Wk : (z == 2) ? Wv : Wo;
  __bf16* Wt = (__bf16*)wt + (size_t)z * Dc * Dc;

  // Coalesced f32 reads; transposed bf16 scatter into LDS.
  #pragma unroll
  for (int i = 0; i < 4; ++i) {
    int idx = tid + i * 256;                   // 1024 elements
    int r = idx >> 5;                          // k within tile
    int c = idx & 31;                          // col within tile
    T[c][r] = (__bf16)W[(size_t)(k0 + r) * Dc + c0 + c];
  }
  __syncthreads();

  // Coalesced dword writes of Wt rows.
  #pragma unroll
  for (int i = 0; i < 2; ++i) {
    int idx = tid + i * 256;                   // dword index, 512 total
    int c = idx >> 4;                          // row of Wt within tile
    int k = (idx & 15) * 2;
    unsigned u = *(const unsigned*)&T[c][k];
    *(unsigned*)(Wt + (size_t)(c0 + c) * Dc + k0 + k) = u;
  }
}

// ---------------------------------------------------------------------------
// Kernel 1: Q/K/V projection (bf16 WMMA, f32 accumulate) with fused RoPE.
// Each wave computes TWO 16x16 output tiles so the A fragment is reused
// across two WMMAs.  B read as contiguous b128 pairs from transposed bf16 W.
// grid = (B*N/16, D/256, 3); block = 256 (8 waves).
// ---------------------------------------------------------------------------
__global__ __launch_bounds__(256)
void qkv_rope_kernel(const float* __restrict__ x,
                     const unsigned short* __restrict__ wt,
                     unsigned short* __restrict__ qo,
                     unsigned short* __restrict__ ko,
                     unsigned short* __restrict__ vo) {
  __shared__ __align__(16) __bf16 Xs[16][1040];  // 16x1024 bf16, padded rows

  const int tid   = threadIdx.x;
  const int lane  = tid & 31;
  const int wv    = tid >> 5;
  const int mtile = blockIdx.x;
  const int z     = blockIdx.z;
  const __bf16* Wt = (const __bf16*)wt + (size_t)z * Dc * Dc;
  __bf16* out      = (__bf16*)((z == 0) ? qo : (z == 1) ? ko : vo);

  // Stage x tile [16 rows x 1024] -> LDS bf16 (coalesced f32x4 loads).
  const float* xrow = x + (size_t)mtile * 16 * Dc;
  #pragma unroll
  for (int i = 0; i < 16; ++i) {
    int idx = tid + i * 256;                 // f32x4 index, 4096 total
    f32x4 f4 = ((const f32x4*)xrow)[idx];
    int r = (idx * 4) >> 10;
    int c = (idx * 4) & (Dc - 1);
    union { u32x2 u; __bf16 h[4]; } pk;
    pk.h[0] = (__bf16)f4[0]; pk.h[1] = (__bf16)f4[1];
    pk.h[2] = (__bf16)f4[2]; pk.h[3] = (__bf16)f4[3];
    *(u32x2*)&Xs[r][c] = pk.u;
  }
  __syncthreads();

  const int ln   = lane & 15;
  const int hl   = lane >> 4;
  const int b8   = hl * 8;
  const int col0 = blockIdx.y * 256 + wv * 32 + ln;
  const __bf16* wrow0 = Wt + (size_t)col0 * Dc + hl * 16;
  const __bf16* wrow1 = wrow0 + (size_t)16 * Dc;

  v8f acc0 = {0.f, 0.f, 0.f, 0.f, 0.f, 0.f, 0.f, 0.f};
  v8f acc1 = acc0;
  #pragma unroll 4
  for (int k0 = 0; k0 < Dc; k0 += 32) {
    v16bf a = load2x8(&Xs[ln][k0 + hl * 16], &Xs[ln][k0 + hl * 16 + 8]);
    const __bf16* bp0 = wrow0 + k0;
    const __bf16* bp1 = wrow1 + k0;
    __builtin_prefetch(bp0 + 128, 0, 1);   // speculative; OOB is dropped
    __builtin_prefetch(bp1 + 128, 0, 1);
    acc0 = wmma_bf16(a, load2x8(bp0, bp0 + 8), acc0);
    acc1 = wmma_bf16(a, load2x8(bp1, bp1 + 8), acc1);
  }

  #pragma unroll
  for (int t = 0; t < 2; ++t) {
    const int col = col0 + t * 16;
    const v8f acc = t ? acc1 : acc0;
    const int hh = col >> 6;        // head
    const int d  = col & (HDc - 1); // dim within head
    #pragma unroll
    for (int r = 0; r < 8; ++r) {
      int M  = r + b8;              // C layout: row = r + 8*(lane/16)
      int g  = mtile * 16 + M;      // global (b,n) row
      int bb = g >> 11;             // / N
      int n  = g & (Nc - 1);
      float vvv = acc[r];
      if (z < 2) {                  // RoPE on Q and K only (z uniform)
        float e    = (float)(d & ~1) * (1.0f / 64.0f);
        float freq = exp2f(-13.287712379549449f * e);   // 10000^-e
        float ang  = (float)n * freq;
        float cs = __cosf(ang), sn = __sinf(ang);
        float partner = __shfl_xor(vvv, 1, 32);         // even<->odd pair
        vvv = (col & 1) ? fmaf(partner, sn, vvv * cs)   // odd:  xe*s + xo*c
                        : fmaf(-partner, sn, vvv * cs); // even: xe*c - xo*s
      }
      out[((size_t)(bb * Hc + hh) * Nc + n) * HDc + d] = (__bf16)vvv;
    }
  }
}

// ---------------------------------------------------------------------------
// Kernel 2: flash attention over one (b,h); block = 8 waves, each a 16-query
// tile; 32 keys per step (8 WMMAs/step), online softmax.  K tile staged by
// the Tensor Data Mover (tensor_load_to_lds) when available; V tile staged
// transposed by the block.
// grid = (B*H, N/128); block = 256.
// ---------------------------------------------------------------------------
__global__ __launch_bounds__(256)
void attn_kernel(const unsigned short* __restrict__ qb,
                 const unsigned short* __restrict__ kb,
                 const unsigned short* __restrict__ vb,
                 unsigned short* __restrict__ obuf) {
  __shared__ __align__(16) __bf16 Kt[32][80];      // [key][hd], 8-dword row pad
  __shared__ __align__(16) __bf16 Vt[64][40];      // transposed: [hd][key]
  __shared__ __align__(16) __bf16 Ps[8][16][40];   // per-wave P staging

  const __bf16* Q  = (const __bf16*)qb;
  const __bf16* Km = (const __bf16*)kb;
  const __bf16* Vm = (const __bf16*)vb;
  __bf16* O        = (__bf16*)obuf;

  const int tid  = threadIdx.x;
  const int lane = tid & 31;
  const int wv   = tid >> 5;
  const int bh   = blockIdx.x;
  const int bb   = bh >> 4;
  const int hh   = bh & 15;
  const int q0   = (blockIdx.y * 8 + wv) * 16;
  const int ln   = lane & 15;
  const int hl   = lane >> 4;
  const int b8   = hl * 8;

  // Q A-fragments for the two 32-wide hd chunks (loaded once).
  const __bf16* qrow = Q + ((size_t)bh * Nc + q0 + ln) * HDc;
  v16bf qa0 = load2x8(qrow + hl * 16,      qrow + hl * 16 + 8);
  v16bf qa1 = load2x8(qrow + 32 + hl * 16, qrow + 40 + hl * 16);

  v8f ot[4];
  float mrow[8], lrow[8];
  const v8f vzero = {0.f, 0.f, 0.f, 0.f, 0.f, 0.f, 0.f, 0.f};
  #pragma unroll
  for (int t = 0; t < 4; ++t) ot[t] = vzero;
  #pragma unroll
  for (int r = 0; r < 8; ++r) { mrow[r] = -3.0e38f; lrow[r] = 0.f; }

  const size_t kvbase = (size_t)bh * Nc * HDc;

  for (int kv0 = 0; kv0 < Nc; kv0 += 32) {
    __syncthreads();  // previous-iteration readers done before restage

#if HAVE_TDM
    if (wv == 0) {
      // D# for a 2D 64x32 bf16 tile -> Kt with 8-dword pad per 32-dword row.
      unsigned lds = (unsigned)(unsigned long long)(uintptr_t)&Kt[0][0];
      unsigned long long ga =
          (unsigned long long)(uintptr_t)(Km + kvbase + (size_t)kv0 * HDc);
      u32x4 g0;
      g0[0] = 1u;                                   // count=1, user descriptor
      g0[1] = lds;                                  // lds_addr (bytes)
      g0[2] = (unsigned)ga;                         // global_addr[31:0]
      g0[3] = (unsigned)((ga >> 32) & 0x1FFFFFFull) | (2u << 30);  // type=2
      i32x8 g1;
      g1[0] = (int)((1u << 16)      // data_size = 2 bytes
                  | (1u << 20)      // pad_enable
                  | (4u << 22)      // pad_interval: 32 dwords
                  | (7u << 25));    // pad_amount:   8 dwords
      g1[1] = (int)((64u & 0xffffu) << 16);          // tensor_dim0 = 64 (lo16)
      g1[2] = (int)((Nc & 0xffffu) << 16);           // dim0 hi=0, tensor_dim1 lo
      g1[3] = (int)(((unsigned)Nc >> 16) | (64u << 16)); // dim1 hi, tile_dim0=64
      g1[4] = (int)(32u);                            // tile_dim1=32, tile_dim2=0
      g1[5] = 64;                                    // tensor_dim0_stride = 64
      g1[6] = 0;
      g1[7] = 0;
      i32x4 zg = {0, 0, 0, 0};
#if __clang_major__ >= 23
      i32x8 zg8 = {0, 0, 0, 0, 0, 0, 0, 0};
      __builtin_amdgcn_tensor_load_to_lds(g0, g1, zg, zg, zg8, 0);
#else
      __builtin_amdgcn_tensor_load_to_lds(g0, g1, zg, zg, 0);
#endif
    }
#endif

    // Cooperative stage of V (transposed).  Overlaps with the TDM K load.
    const unsigned int* vg =
        (const unsigned int*)(Vm + kvbase + (size_t)kv0 * HDc);
#if !HAVE_TDM
    const unsigned int* kg =
        (const unsigned int*)(Km + kvbase + (size_t)kv0 * HDc);
#endif
    #pragma unroll
    for (int i = 0; i < 4; ++i) {
      int idx = tid + i * 256;                 // dword index, 1024 total
      int key = idx >> 5;
      int hd  = (idx & 31) * 2;
      union { unsigned int u; __bf16 h[2]; } s;
      s.u = vg[idx];
      Vt[hd][key]     = s.h[0];
      Vt[hd + 1][key] = s.h[1];
#if !HAVE_TDM
      *(unsigned int*)&Kt[key][hd] = kg[idx];
#endif
    }

    // Speculative prefetch of next K/V tiles (4KB each, spread over block).
    __builtin_prefetch(Vm + kvbase + (size_t)(kv0 + 32) * HDc + tid * 8, 0, 1);
    __builtin_prefetch(Km + kvbase + (size_t)(kv0 + 32) * HDc + tid * 8, 0, 1);

#if HAVE_TDM && __has_builtin(__builtin_amdgcn_s_wait_tensorcnt)
    if (wv == 0) __builtin_amdgcn_s_wait_tensorcnt(0);
#endif
    __syncthreads();

    // S = Q K^T  (two 16-key tiles, hd split 2x32).
    v8f s0 = vzero, s1 = vzero;
    {
      const __bf16* kr0 = &Kt[ln][hl * 16];
      s0 = wmma_bf16(qa0, load2x8(kr0,      kr0 + 8),  s0);
      s0 = wmma_bf16(qa1, load2x8(kr0 + 32, kr0 + 40), s0);
      const __bf16* kr1 = &Kt[16 + ln][hl * 16];
      s1 = wmma_bf16(qa0, load2x8(kr1,      kr1 + 8),  s1);
      s1 = wmma_bf16(qa1, load2x8(kr1 + 32, kr1 + 40), s1);
    }

    // Online softmax per row (rows = lane-half x 8 VGPRs).
    #pragma unroll
    for (int r = 0; r < 8; ++r) {
      float a0 = s0[r] * 0.125f;               // hd^-0.5 = 1/8
      float a1 = s1[r] * 0.125f;
      float mx = fmaxf(a0, a1);
      mx = fmaxf(mx, __shfl_xor(mx, 1, 32));
      mx = fmaxf(mx, __shfl_xor(mx, 2, 32));
      mx = fmaxf(mx, __shfl_xor(mx, 4, 32));
      mx = fmaxf(mx, __shfl_xor(mx, 8, 32));
      float mn = fmaxf(mrow[r], mx);
      float p0 = exp2f((a0 - mn) * 1.4426950408889634f);
      float p1 = exp2f((a1 - mn) * 1.4426950408889634f);
      float rs = p0 + p1;
      rs += __shfl_xor(rs, 1, 32);
      rs += __shfl_xor(rs, 2, 32);
      rs += __shfl_xor(rs, 4, 32);
      rs += __shfl_xor(rs, 8, 32);
      float alpha = exp2f((mrow[r] - mn) * 1.4426950408889634f);
      lrow[r] = lrow[r] * alpha + rs;
      mrow[r] = mn;
      ot[0][r] *= alpha; ot[1][r] *= alpha; ot[2][r] *= alpha; ot[3][r] *= alpha;
      int M = r + b8;
      Ps[wv][M][ln]      = (__bf16)p0;         // C->A layout via LDS
      Ps[wv][M][16 + ln] = (__bf16)p1;
    }

    // P (16x32) A-fragment; per-wave LDS, DS ops stay in order within a wave.
    const __bf16* pr = &Ps[wv][ln][0];
    v16bf pa = load2x8(pr + hl * 16, pr + hl * 16 + 8);
    #pragma unroll
    for (int t = 0; t < 4; ++t) {
      const __bf16* vr = &Vt[t * 16 + ln][hl * 16];
      ot[t] = wmma_bf16(pa, load2x8(vr, vr + 8), ot[t]);
    }
  }

  // Normalize and store O as bf16 in token-major [B*N, D] layout.
  #pragma unroll
  for (int r = 0; r < 8; ++r) {
    float inv = 1.0f / lrow[r];
    int M = r + b8;
    size_t rowoff = ((size_t)bb * Nc + q0 + M) * Dc + hh * HDc;
    #pragma unroll
    for (int t = 0; t < 4; ++t)
      O[rowoff + t * 16 + ln] = (__bf16)(ot[t][r] * inv);
  }
}

// ---------------------------------------------------------------------------
// Kernel 3: output projection  out = O(bf16) x Wo -> f32 (Wo pre-transposed).
// Two 16x16 tiles per wave (A fragment reused).
// grid = (B*N/16, D/256); block = 256.
// ---------------------------------------------------------------------------
__global__ __launch_bounds__(256)
void oproj_kernel(const unsigned short* __restrict__ ob,
                  const unsigned short* __restrict__ wt,
                  float* __restrict__ out) {
  __shared__ __align__(16) __bf16 As[16][1040];

  const int tid   = threadIdx.x;
  const int lane  = tid & 31;
  const int wv    = tid >> 5;
  const int mtile = blockIdx.x;
  const __bf16* Wt = (const __bf16*)wt + (size_t)3 * Dc * Dc;  // Wo^T

  const __bf16* Ob = (const __bf16*)ob;
  const u32x4* src = (const u32x4*)(Ob + (size_t)mtile * 16 * Dc);
  #pragma unroll
  for (int i = 0; i < 8; ++i) {
    int idx = tid + i * 256;                   // u32x4 index, 2048 total
    int r = (idx * 8) >> 10;
    int c = (idx * 8) & (Dc - 1);
    *(u32x4*)&As[r][c] = src[idx];
  }
  __syncthreads();

  const int ln   = lane & 15;
  const int hl   = lane >> 4;
  const int b8   = hl * 8;
  const int col0 = blockIdx.y * 256 + wv * 32 + ln;
  const __bf16* wrow0 = Wt + (size_t)col0 * Dc + hl * 16;
  const __bf16* wrow1 = wrow0 + (size_t)16 * Dc;

  v8f acc0 = {0.f, 0.f, 0.f, 0.f, 0.f, 0.f, 0.f, 0.f};
  v8f acc1 = acc0;
  #pragma unroll 4
  for (int k0 = 0; k0 < Dc; k0 += 32) {
    v16bf a = load2x8(&As[ln][k0 + hl * 16], &As[ln][k0 + hl * 16 + 8]);
    const __bf16* bp0 = wrow0 + k0;
    const __bf16* bp1 = wrow1 + k0;
    __builtin_prefetch(bp0 + 128, 0, 1);
    __builtin_prefetch(bp1 + 128, 0, 1);
    acc0 = wmma_bf16(a, load2x8(bp0, bp0 + 8), acc0);
    acc1 = wmma_bf16(a, load2x8(bp1, bp1 + 8), acc1);
  }

  #pragma unroll
  for (int r = 0; r < 8; ++r) {
    int g = mtile * 16 + r + b8;
    out[(size_t)g * Dc + col0]      = acc0[r];
    out[(size_t)g * Dc + col0 + 16] = acc1[r];
  }
}

// ---------------------------------------------------------------------------
extern "C" void kernel_launch(void* const* d_in, const int* in_sizes, int n_in,
                              void* d_out, int out_size, void* d_ws, size_t ws_size,
                              hipStream_t stream) {
  (void)in_sizes; (void)n_in; (void)out_size; (void)ws_size;
  const float* x  = (const float*)d_in[0];
  const float* Wq = (const float*)d_in[1];
  const float* Wk = (const float*)d_in[2];
  const float* Wv = (const float*)d_in[3];
  const float* Wo = (const float*)d_in[4];

  char* ws = (char*)d_ws;
  const size_t seg = (size_t)Bc * Hc * Nc * HDc * sizeof(unsigned short); // 8 MiB
  unsigned short* Qb = (unsigned short*)(ws + 0 * seg);
  unsigned short* Kb = (unsigned short*)(ws + 1 * seg);
  unsigned short* Vb = (unsigned short*)(ws + 2 * seg);
  unsigned short* Ob = (unsigned short*)(ws + 3 * seg);
  unsigned short* Wt = (unsigned short*)(ws + 4 * seg);  // 4 x 2 MiB bf16 W^T

  wtrans_kernel<<<dim3(Dc / 32, Dc / 32, 4), 256, 0, stream>>>(Wq, Wk, Wv, Wo, Wt);
  qkv_rope_kernel<<<dim3(Bc * Nc / 16, Dc / 256, 3), 256, 0, stream>>>(
      x, Wt, Qb, Kb, Vb);
  attn_kernel<<<dim3(Bc * Hc, Nc / 128), 256, 0, stream>>>(Qb, Kb, Vb, Ob);
  oproj_kernel<<<dim3(Bc * Nc / 16, Dc / 256), 256, 0, stream>>>(
      Ob, Wt, (float*)d_out);
}
